// FCOS_30270929502579
// MI455X (gfx1250) — matrix operand order repaired and checked by reading, not compiled, not run on hardware
//
#include <hip/hip_runtime.h>
#include <hip/hip_bf16.h>

typedef __bf16 bf16;
typedef __attribute__((ext_vector_type(16))) __bf16 v16bf;
typedef __attribute__((ext_vector_type(8)))  __bf16 v8bf;
typedef __attribute__((ext_vector_type(8)))  float  v8f;

// ---------------------------------------------------------------------------
// NCHW fp32 -> NHWC bf16 conversion (dst[b][n][c] = src[b][c][n])
// ---------------------------------------------------------------------------
__global__ void nchw2nhwc_kernel(const float* __restrict__ src, bf16* __restrict__ dst,
                                 int C, int N, int total) {
  int idx = blockIdx.x * blockDim.x + threadIdx.x;
  if (idx >= total) return;
  int c = idx % C;
  int n = (idx / C) % N;
  int b = idx / (C * N);
  dst[idx] = (bf16)src[((size_t)b * C + c) * N + n];
}

// ---------------------------------------------------------------------------
// Pack conv weights (Cout, Cin, kk) fp32 -> (Mpad, Kpad) bf16 with K reordered
// as k = r*Cin + ci  (r = spatial tap dy*ks+dx). Zero pad in M and K.
// ---------------------------------------------------------------------------
__global__ void pack_w_kernel(const float* __restrict__ w, bf16* __restrict__ dst,
                              int Cout, int Cin, int kk, int Kpad, int Mpad) {
  int idx = blockIdx.x * blockDim.x + threadIdx.x;
  if (idx >= Mpad * Kpad) return;
  int m = idx / Kpad, k = idx % Kpad;
  float v = 0.f;
  int K = Cin * kk;
  if (m < Cout && k < K) {
    int r = k / Cin, ci = k - r * Cin;
    v = w[((size_t)m * Cin + ci) * kk + r];
  }
  dst[idx] = (bf16)v;
}

// Merged box(4ch) + ctr(1ch) head weights -> rows 0..3 box, row 4 ctr.
__global__ void pack_bc_w_kernel(const float* __restrict__ bw, const float* __restrict__ cw,
                                 bf16* __restrict__ dst, int Cin, int kk, int Kpad, int Mpad) {
  int idx = blockIdx.x * blockDim.x + threadIdx.x;
  if (idx >= Mpad * Kpad) return;
  int m = idx / Kpad, k = idx % Kpad;
  float v = 0.f;
  int K = Cin * kk;
  if (k < K) {
    int r = k / Cin, ci = k - r * Cin;
    if (m < 4)       v = bw[((size_t)m * Cin + ci) * kk + r];
    else if (m == 4) v = cw[((size_t)ci) * kk + r];
  }
  dst[idx] = (bf16)v;
}

__global__ void bias_bc_kernel(const float* __restrict__ bw, const float* __restrict__ cw,
                               float* __restrict__ dst) {
  int i = threadIdx.x;
  if (i < 4) dst[i] = bw[i];
  else if (i == 4) dst[i] = cw[0];
}

// ---------------------------------------------------------------------------
// Nearest-neighbor 2x upsample-add, NHWC, C=256 (in place on hi-res lateral).
// ---------------------------------------------------------------------------
__global__ void upsample_add_kernel(bf16* __restrict__ hi, const bf16* __restrict__ lo,
                                    int total, int H, int W) {
  int idx = blockIdx.x * blockDim.x + threadIdx.x;
  if (idx >= total) return;
  int c = idx & 255;
  int p = idx >> 8;
  int x = p % W;
  int y = (p / W) % H;
  int b = p / (W * H);
  size_t li = ((((size_t)b * (H >> 1) + (y >> 1)) * (W >> 1) + (x >> 1)) << 8) | c;
  hi[idx] = (bf16)((float)hi[idx] + (float)lo[li]);
}

// --- async global->LDS helpers (CDNA5 ASYNC path, GV addressing mode) ------
__device__ __forceinline__ void async_b128(const void* gptr, void* lds_ptr) {
  unsigned       l = (unsigned)(uintptr_t)lds_ptr;          // LDS aperture: addr[31:0]
  unsigned long long g = (unsigned long long)(uintptr_t)gptr;
  asm volatile("global_load_async_to_lds_b128 %0, %1, off"
               :: "v"(l), "v"(g) : "memory");
}
__device__ __forceinline__ void wait_async0() {
  asm volatile("s_wait_asynccnt 0" ::: "memory");
}

// ---------------------------------------------------------------------------
// Implicit-GEMM conv via v_wmma_f32_16x16x32_bf16, NHWC activations.
//   KS=3: Cin=256, K=Kpad=2304 compile-time; every 32-wide K slab shares one
//         spatial tap -> contiguous 16B async LDS-DMA staging loads.
//   KS=1: runtime Cin/Kpad (laterals); async weights, vector/tail activations.
//   Workgroup: 256 thr = 8 waves; tile 128(M) x 128(N); wave = 32x64 = 8 WMMA.
//   Double-buffered LDS; next slab staged (async DMA) while WMMAs consume.
// ---------------------------------------------------------------------------
template <int KS>
__global__ __launch_bounds__(256)
void conv_wmma_kernel(const bf16* __restrict__ in, const bf16* __restrict__ wp,
                      const float* __restrict__ bias,
                      bf16* __restrict__ out_act, float* __restrict__ out_head,
                      int Cin_rt, int H, int W,
                      int Cout, int Kpad_rt, int relu, int co_base, int loc_base) {
  const int Cin  = (KS == 3) ? 256  : Cin_rt;
  const int K    = Cin * KS * KS;
  const int Kpad = (KS == 3) ? 2304 : Kpad_rt;
  const int N    = H * W;
  const int n0   = blockIdx.x * 128;
  const int m0   = blockIdx.y * 128;
  const int b    = blockIdx.z;
  const bf16* __restrict__ inb = in + (size_t)b * N * Cin;

  __shared__ __align__(32) bf16 Alds[2][128 * 32];
  __shared__ __align__(32) bf16 Blds[2][128 * 32];

  const int tid   = threadIdx.x;
  const int lane  = tid & 31;
  const int wave  = tid >> 5;
  const int waveM = (wave & 3) * 32;   // 4 waves along M
  const int waveN = (wave >> 2) * 64;  // 2 waves along N

  v8f acc[2][4] = {};

  // staging indices: 128 rows x 32 k, each thread moves 16 contiguous k (32B)
  const int arow = tid >> 1, akh = (tid & 1) * 16;
  const int bn   = tid >> 1, bkh = (tid & 1) * 16;
  const int gy   = (n0 + bn) / W;
  const int gx   = (n0 + bn) % W;

  const int nslabs = Kpad / 32;

  auto stageA = [&](int buf, int k0) {
    const bf16* src = wp + (size_t)(m0 + arow) * Kpad + k0 + akh;
    bf16* dst = &Alds[buf][arow * 32 + akh];
    async_b128(src,     dst);
    async_b128(src + 8, dst + 8);
  };
  auto stageB = [&](int buf, int k0) {
    bf16* dst = &Blds[buf][bn * 32 + bkh];
    if (KS == 3) {
      const int r   = k0 >> 8;              // spatial tap, constant over slab
      const int ci0 = (k0 & 255) + bkh;
      const int dy  = r / 3, dx = r - 3 * (r / 3);
      const int yy  = gy + dy - 1, xx = gx + dx - 1;
      if (yy >= 0 && yy < H && xx >= 0 && xx < W) {
        const bf16* s = inb + ((size_t)(yy * W + xx) << 8) + ci0;
        async_b128(s,     dst);
        async_b128(s + 8, dst + 8);
      } else {
        v8bf z = {};
        *(v8bf*)dst       = z;              // zero-fill padding taps via DS path
        *(v8bf*)(dst + 8) = z;
      }
    } else {
      const int k = k0 + bkh;
      const bf16* base = inb + (size_t)(n0 + bn) * Cin;
      if (k + 16 <= K) {
        async_b128(base + k,     dst);
        async_b128(base + k + 8, dst + 8);
      } else {
        v8bf v0 = {}, v1 = {};
        #pragma unroll
        for (int j = 0; j < 8; ++j) v0[j] = (k + j     < K) ? base[k + j]     : (bf16)0.f;
        #pragma unroll
        for (int j = 0; j < 8; ++j) v1[j] = (k + 8 + j < K) ? base[k + 8 + j] : (bf16)0.f;
        *(v8bf*)dst       = v0;
        *(v8bf*)(dst + 8) = v1;
      }
    }
  };

  stageA(0, 0);
  stageB(0, 0);
  wait_async0();
  __syncthreads();

  for (int s = 0; s < nslabs; ++s) {
    const int buf = s & 1;
    if (s + 1 < nslabs) {
      stageA(buf ^ 1, (s + 1) * 32);       // async DMA overlaps the WMMAs below
      stageB(buf ^ 1, (s + 1) * 32);
    }
    if (s + 2 < nslabs)                    // keep weight stream hot in L2
      __builtin_prefetch(wp + (size_t)(m0 + arow) * Kpad + (s + 2) * 32, 0, 0);

    // fragments in contiguous k order (k = 16*(lane>>4) + s), same for A and B
    const int halfk = (lane >> 4) * 16;
    v16bf afrag[2], bfrag[4];
    #pragma unroll
    for (int i = 0; i < 2; ++i) {
      const bf16* ar = &Alds[buf][(waveM + i * 16 + (lane & 15)) * 32];
      union { v16bf v; v8bf h[2]; } u;
      u.h[0] = *(const v8bf*)(ar + halfk);
      u.h[1] = *(const v8bf*)(ar + halfk + 8);
      afrag[i] = u.v;
    }
    #pragma unroll
    for (int j = 0; j < 4; ++j) {
      const bf16* br = &Blds[buf][(waveN + j * 16 + (lane & 15)) * 32 + (lane >> 4) * 16];
      bfrag[j] = *(const v16bf*)br;
    }
    #pragma unroll
    for (int i = 0; i < 2; ++i)
      #pragma unroll
      for (int j = 0; j < 4; ++j)
        acc[i][j] = __builtin_amdgcn_wmma_f32_16x16x32_bf16(
            false, afrag[i], false, bfrag[j], (short)0, acc[i][j], false, false);

    wait_async0();                          // published buffer complete
    __syncthreads();
  }

  // ---- epilogue: C layout VGPR r -> M = r + 8*(lane>=16), N = lane&15 ----
  #pragma unroll
  for (int i = 0; i < 2; ++i) {
    const int mb = m0 + waveM + i * 16 + (lane >> 4) * 8;   // 8 consecutive channels
    #pragma unroll
    for (int j = 0; j < 4; ++j) {
      const int nn = n0 + waveN + j * 16 + (lane & 15);
      if (out_head) {
        float* o = out_head + ((size_t)(b * 5376 + loc_base + nn)) * 85 + co_base + mb;
        #pragma unroll
        for (int r = 0; r < 8; ++r)
          if (mb + r < Cout) o[r] = acc[i][j][r] + bias[mb + r];
      } else {
        v8bf ov;
        #pragma unroll
        for (int r = 0; r < 8; ++r) {
          float v = acc[i][j][r] + bias[mb + r];
          if (relu) v = fmaxf(v, 0.f);
          ov[r] = (bf16)v;
        }
        *(v8bf*)&out_act[((size_t)b * N + nn) * Cout + mb] = ov;   // NHWC, 16B store
      }
    }
  }
}

// ---------------------------------------------------------------------------
// Host orchestration
// ---------------------------------------------------------------------------
extern "C" void kernel_launch(void* const* d_in, const int* in_sizes, int n_in,
                              void* d_out, int out_size, void* d_ws, size_t ws_size,
                              hipStream_t stream) {
  (void)in_sizes; (void)n_in; (void)out_size; (void)ws_size;
  const int B = 2;

  const float* c3     = (const float*)d_in[0];
  const float* c4     = (const float*)d_in[1];
  const float* c5     = (const float*)d_in[2];
  const float* lat3_w = (const float*)d_in[3];
  const float* lat3_b = (const float*)d_in[4];
  const float* out3_w = (const float*)d_in[5];
  const float* out3_b = (const float*)d_in[6];
  const float* lat4_w = (const float*)d_in[7];
  const float* lat4_b = (const float*)d_in[8];
  const float* out4_w = (const float*)d_in[9];
  const float* out4_b = (const float*)d_in[10];
  const float* lat5_w = (const float*)d_in[11];
  const float* lat5_b = (const float*)d_in[12];
  const float* out5_w = (const float*)d_in[13];
  const float* out5_b = (const float*)d_in[14];
  const float* scw    = (const float*)d_in[15];
  const float* scb    = (const float*)d_in[16];
  const float* sbw    = (const float*)d_in[17];
  const float* sbb    = (const float*)d_in[18];
  const float* cls_w  = (const float*)d_in[19];
  const float* cls_b  = (const float*)d_in[20];
  const float* box_w  = (const float*)d_in[21];
  const float* box_b  = (const float*)d_in[22];
  const float* ctr_w  = (const float*)d_in[23];
  const float* ctr_b  = (const float*)d_in[24];
  float* out = (float*)d_out;

  uintptr_t base = (uintptr_t)d_ws;
  size_t off = 0;
  auto alloc = [&](size_t bytes) -> void* {
    off = (off + 255) & ~(size_t)255;
    void* p = (void*)(base + off);
    off += bytes;
    return p;
  };
  auto allocBf = [&](size_t n) -> bf16*  { return (bf16*)alloc(n * sizeof(bf16)); };
  auto allocF  = [&](size_t n) -> float* { return (float*)alloc(n * sizeof(float)); };

  // activations (bf16, NHWC)
  bf16* c3b = allocBf((size_t)B * 4096 * 64);
  bf16* c4b = allocBf((size_t)B * 1024 * 160);
  bf16* c5b = allocBf((size_t)B * 256  * 400);
  bf16* l3  = allocBf((size_t)B * 4096 * 256);
  bf16* l4  = allocBf((size_t)B * 1024 * 256);
  bf16* l5  = allocBf((size_t)B * 256  * 256);
  bf16* p3  = allocBf((size_t)B * 4096 * 256);
  bf16* p4  = allocBf((size_t)B * 1024 * 256);
  bf16* p5  = allocBf((size_t)B * 256  * 256);
  bf16* sA  = allocBf((size_t)B * 4096 * 256);
  bf16* sB  = allocBf((size_t)B * 4096 * 256);

  // packed weights (bf16, [Mpad][Kpad], K-order (tap, ci))
  bf16* wlat3 = allocBf((size_t)256 * 64);
  bf16* wlat4 = allocBf((size_t)256 * 160);
  bf16* wlat5 = allocBf((size_t)256 * 416);
  bf16* wout3 = allocBf((size_t)256 * 2304);
  bf16* wout4 = allocBf((size_t)256 * 2304);
  bf16* wout5 = allocBf((size_t)256 * 2304);
  bf16 *wsc[4], *wsb[4];
  for (int l = 0; l < 4; ++l) wsc[l] = allocBf((size_t)256 * 2304);
  for (int l = 0; l < 4; ++l) wsb[l] = allocBf((size_t)256 * 2304);
  bf16* whcls = allocBf((size_t)128 * 2304);
  bf16* whbc  = allocBf((size_t)128 * 2304);
  float* bbc  = allocF(8);

  // --- NCHW->NHWC conversions ---
  auto cvt = [&](const float* s, bf16* d, int C, int N) {
    int n = B * C * N;
    nchw2nhwc_kernel<<<(n + 255) / 256, 256, 0, stream>>>(s, d, C, N, n);
  };
  cvt(c3, c3b, 64, 4096);
  cvt(c4, c4b, 160, 1024);
  cvt(c5, c5b, 400, 256);

  // --- weight packing ---
  auto pack = [&](const float* w, bf16* d, int Cout, int Cin, int kk, int Kpad, int Mpad) {
    int n = Mpad * Kpad;
    pack_w_kernel<<<(n + 255) / 256, 256, 0, stream>>>(w, d, Cout, Cin, kk, Kpad, Mpad);
  };
  pack(lat3_w, wlat3, 256, 64,  1, 64,  256);
  pack(lat4_w, wlat4, 256, 160, 1, 160, 256);
  pack(lat5_w, wlat5, 256, 400, 1, 416, 256);
  pack(out3_w, wout3, 256, 256, 9, 2304, 256);
  pack(out4_w, wout4, 256, 256, 9, 2304, 256);
  pack(out5_w, wout5, 256, 256, 9, 2304, 256);
  for (int l = 0; l < 4; ++l) pack(scw + (size_t)l * 256 * 256 * 9, wsc[l], 256, 256, 9, 2304, 256);
  for (int l = 0; l < 4; ++l) pack(sbw + (size_t)l * 256 * 256 * 9, wsb[l], 256, 256, 9, 2304, 256);
  pack(cls_w, whcls, 80, 256, 9, 2304, 128);
  {
    int n = 128 * 2304;
    pack_bc_w_kernel<<<(n + 255) / 256, 256, 0, stream>>>(box_w, ctr_w, whbc, 256, 9, 2304, 128);
  }
  bias_bc_kernel<<<1, 8, 0, stream>>>(box_b, ctr_b, bbc);

  // --- conv launchers ---
  auto conv3 = [&](const bf16* in, const bf16* wp, const float* bias,
                   bf16* oact, float* ohead, int H, int W,
                   int Cout, int Mpad, int relu, int co_base, int loc_base) {
    dim3 g((H * W) / 128, Mpad / 128, B);
    conv_wmma_kernel<3><<<g, 256, 0, stream>>>(in, wp, bias, oact, ohead,
        256, H, W, Cout, 2304, relu, co_base, loc_base);
  };
  auto conv1 = [&](const bf16* in, const bf16* wp, const float* bias,
                   bf16* oact, int Cin, int H, int W, int Kpad) {
    dim3 g((H * W) / 128, 2, B);
    conv_wmma_kernel<1><<<g, 256, 0, stream>>>(in, wp, bias, oact, nullptr,
        Cin, H, W, 256, Kpad, 0, 0, 0);
  };

  // --- FPN ---
  conv1(c3b, wlat3, lat3_b, l3, 64,  64, 64, 64);
  conv1(c4b, wlat4, lat4_b, l4, 160, 32, 32, 160);
  conv1(c5b, wlat5, lat5_b, l5, 400, 16, 16, 416);
  conv3(l5, wout5, out5_b, p5, nullptr, 16, 16, 256, 256, 0, 0, 0);
  {
    int n = B * 256 * 1024;
    upsample_add_kernel<<<(n + 255) / 256, 256, 0, stream>>>(l4, l5, n, 32, 32);
  }
  conv3(l4, wout4, out4_b, p4, nullptr, 32, 32, 256, 256, 0, 0, 0);
  {
    int n = B * 256 * 4096;
    upsample_add_kernel<<<(n + 255) / 256, 256, 0, stream>>>(l3, l4, n, 64, 64);
  }
  conv3(l3, wout3, out3_b, p3, nullptr, 64, 64, 256, 256, 0, 0, 0);

  // --- heads per level ---
  struct Lvl { bf16* p; int H, W, loc; };
  Lvl lv[3] = { {p3, 64, 64, 0}, {p4, 32, 32, 4096}, {p5, 16, 16, 5120} };
  for (int li = 0; li < 3; ++li) {
    const Lvl& L = lv[li];
    const bf16* cur = L.p;
    for (int l = 0; l < 4; ++l) {
      bf16* dst = (l & 1) ? sB : sA;
      conv3(cur, wsc[l], scb + l * 256, dst, nullptr, L.H, L.W, 256, 256, 1, 0, 0);
      cur = dst;
    }
    conv3(cur, whcls, cls_b, nullptr, out, L.H, L.W, 80, 128, 0, 0, L.loc);
    cur = L.p;
    for (int l = 0; l < 4; ++l) {
      bf16* dst = (l & 1) ? sB : sA;
      conv3(cur, wsb[l], sbb + l * 256, dst, nullptr, L.H, L.W, 256, 256, 1, 0, 0);
      cur = dst;
    }
    conv3(cur, whbc, bbc, nullptr, out, L.H, L.W, 5, 128, 0, 80, L.loc);
  }
}
